// up_conv_dcoa_31198642438314
// MI455X (gfx1250) — matrix operand order repaired and checked by reading, not compiled, not run on hardware
//
#include <hip/hip_runtime.h>
#include <hip/hip_bf16.h>

typedef __attribute__((ext_vector_type(16))) __bf16        v16bf;
typedef __attribute__((ext_vector_type(8)))  float         v8f;
typedef __attribute__((ext_vector_type(8)))  unsigned int  v8u;
typedef __attribute__((ext_vector_type(4)))  unsigned int  v4u;
typedef __attribute__((ext_vector_type(8)))  int           v8i;
typedef __attribute__((ext_vector_type(4)))  int           v4i;

#define PITCH 34   // halfwords per x position in LDS input tile (32 ch + 2 pad, odd dword stride)

#if __has_builtin(__builtin_amdgcn_tensor_load_to_lds) && __has_builtin(__builtin_amdgcn_s_wait_tensorcnt)
#define USE_TDM 1
#else
#define USE_TDM 0
#endif

__device__ __forceinline__ unsigned short f32_to_bf16(float f) {
  unsigned int u = __float_as_uint(f);
  u += 0x7FFFu + ((u >> 16) & 1u);   // round-to-nearest-even
  return (unsigned short)(u >> 16);
}

// ---------------------------------------------------------------------------
// Kernel 1: pooled[b*32+ci] = mean over 256x256 of x[b,ci,:,:]
// (mean of nearest-2x-upsampled tensor == mean of original tensor)
// ---------------------------------------------------------------------------
__global__ __launch_bounds__(256) void k_pool(const float* __restrict__ x,
                                              float* __restrict__ pooled) {
  __shared__ float red[256];
  const float* p = x + (size_t)blockIdx.x * 65536u;
  float s = 0.f;
  for (int i = threadIdx.x; i < 65536; i += 256) s += p[i];
  red[threadIdx.x] = s;
  __syncthreads();
  for (int off = 128; off > 0; off >>= 1) {
    if ((int)threadIdx.x < off) red[threadIdx.x] += red[threadIdx.x + off];
    __syncthreads();
  }
  if (threadIdx.x == 0) pooled[blockIdx.x] = red[0] * (1.f / 65536.f);
}

// ---------------------------------------------------------------------------
// Kernel 2: per-batch attention softmax(logits/34), aggregate K=4 kernels,
// fold BN scale into bf16 weights, fold bias+BN shift into beta2[b][co].
// aggW layout: [b][tap(9)][co(16)][ci(32)] bf16
// ---------------------------------------------------------------------------
__global__ __launch_bounds__(256) void k_attn_agg(
    const float* __restrict__ pooled, const float* __restrict__ weight,
    const float* __restrict__ bias,   const float* __restrict__ fc1_w,
    const float* __restrict__ fc2_w,  const float* __restrict__ fc2_b,
    const float* __restrict__ gamma,  const float* __restrict__ beta,
    const float* __restrict__ mean,   const float* __restrict__ var,
    unsigned short* __restrict__ aggW, float* __restrict__ beta2) {
  __shared__ float attnS[4];
  __shared__ float scaleS[16];
  __shared__ float shiftS[16];
  const int b = blockIdx.x, tid = threadIdx.x;
  if (tid == 0) {
    float h[9];
    #pragma unroll
    for (int j = 0; j < 9; ++j) {
      float s = 0.f;
      for (int ci = 0; ci < 32; ++ci) s += pooled[b * 32 + ci] * fc1_w[j * 32 + ci];
      h[j] = s > 0.f ? s : 0.f;
    }
    float lg[4], mx = -1e30f;
    #pragma unroll
    for (int k = 0; k < 4; ++k) {
      float s = fc2_b[k];
      for (int j = 0; j < 9; ++j) s += h[j] * fc2_w[k * 9 + j];
      lg[k] = s * (1.0f / 34.0f);            // temperature
      mx = lg[k] > mx ? lg[k] : mx;
    }
    float den = 0.f;
    #pragma unroll
    for (int k = 0; k < 4; ++k) { lg[k] = __expf(lg[k] - mx); den += lg[k]; }
    #pragma unroll
    for (int k = 0; k < 4; ++k) attnS[k] = lg[k] / den;
  }
  if (tid < 16) {
    float sc = gamma[tid] * __frsqrt_rn(var[tid] + 1e-5f);
    scaleS[tid] = sc;
    shiftS[tid] = beta[tid] - mean[tid] * sc;
  }
  __syncthreads();
  if (tid < 16) {
    float s = 0.f;
    #pragma unroll
    for (int k = 0; k < 4; ++k) s += attnS[k] * bias[k * 16 + tid];
    beta2[b * 16 + tid] = s * scaleS[tid] + shiftS[tid];
  }
  for (int idx = tid; idx < 9 * 16 * 32; idx += 256) {
    const int t9 = idx >> 9;            // tap
    const int co = (idx >> 5) & 15;
    float s = 0.f;
    #pragma unroll
    for (int k = 0; k < 4; ++k)
      s += attnS[k] * weight[(((k * 16 + co) * 32 + (idx & 31)) * 9) + t9];
    aggW[b * 4608 + idx] = f32_to_bf16(s * scaleS[co]);
  }
}

// ---------------------------------------------------------------------------
// Kernel 3: fused nearest-upsample + per-sample 3x3 conv (implicit GEMM via
// v_wmma_f32_16x16x32_bf16) + bias/BN + ReLU.
// Block = 256 thr (8 waves), batch b, output rows y0..y0+1.
// Wave tile: D[16co x 16pix]; K = 9 taps x 32 channels -> 9 WMMAs per tile.
// Weights staged into LDS by the Tensor Data Mover (TENSORcnt).
// ---------------------------------------------------------------------------
__global__ __launch_bounds__(256) void k_conv_wmma(
    const float* __restrict__ x,              // [8,32,256,256]
    const unsigned short* __restrict__ aggW,  // bf16 [8][9][16][32]
    const float* __restrict__ beta2,          // [8][16]
    float* __restrict__ out) {                // [8,16,512,512]
  __shared__ unsigned short sbuf[3 * 256 * PITCH]; // 3 src rows, [row][x][ci], 52224 B
  __shared__ unsigned short wbuf[9 * 16 * 32];     // [tap][co][ci], 9216 B

  const int y0   = (int)blockIdx.x * 2;  // even output row
  const int b    = (int)blockIdx.y;
  const int tid  = (int)threadIdx.x;
  const int lane = tid & 31;
  const int col  = lane & 15;            // N column / M row within tile
  const int grp  = lane >> 4;            // lane half
  const int wave = tid >> 5;

  // ---- stage aggregated weights (9216 B contiguous) via TDM ----
#if USE_TDM
  if (wave == 0) {
    const unsigned lds_off = (unsigned)(uintptr_t)(&wbuf[0]);   // generic LDS addr[31:0] = LDS offset
    const unsigned long long ga = (unsigned long long)(const void*)(aggW + b * 4608);
    v4u g0;
    g0[0] = 1u;                                           // count=1 (valid), user mode
    g0[1] = lds_off;                                      // lds_addr
    g0[2] = (unsigned)(ga & 0xffffffffu);                 // global_addr[31:0]
    g0[3] = (unsigned)((ga >> 32) & 0x01ffffffu) | (2u << 30); // global_addr[56:32] | type=2
    v8i g1;
    g1[0] = 0x30000;                    // workgroup_mask=0, data_size=3 (8B), no pad/iter
    g1[1] = (int)(1152u << 16);         // tensor_dim0[15:0]=1152 in bits[63:48]
    g1[2] = (int)(1u << 16);            // tensor_dim0 hi=0 | tensor_dim1=1
    g1[3] = (int)(1152u << 16);         // tensor_dim1 hi=0 | tile_dim0=1152
    g1[4] = 0;                          // tile_dim1=0 (unused), tile_dim2=0
    g1[5] = 1152;                       // tensor_dim0_stride[31:0]
    g1[6] = 0;
    g1[7] = 0;
    v4i gz = {0, 0, 0, 0};
#if __clang_major__ >= 23
    v8i gz8 = {0, 0, 0, 0, 0, 0, 0, 0};
    __builtin_amdgcn_tensor_load_to_lds(g0, g1, gz, gz, gz8, 0);
#else
    __builtin_amdgcn_tensor_load_to_lds(g0, g1, gz, gz, 0);
#endif
    __builtin_amdgcn_s_wait_tensorcnt(0);
  }
#else
  {
    const unsigned int* src = (const unsigned int*)(aggW + b * 4608);
    unsigned int* dst = (unsigned int*)wbuf;
    for (int i = tid; i < 2304; i += 256) dst[i] = src[i];
  }
#endif

  // ---- stage 3 source rows (fp32 -> bf16), coalesced global, conflict-free LDS ----
  const int rbase = (y0 >> 1) - 1;
  for (int i = tid; i < 3 * 32 * 256; i += 256) {
    const int xp  = i & 255;
    const int ci  = (i >> 8) & 31;
    const int row = i >> 13;             // 0..2
    int ys = rbase + row;
    ys = ys < 0 ? 0 : (ys > 255 ? 255 : ys);
    const float v = x[((size_t)(b * 32 + ci) * 256 + ys) * 256 + xp];
    sbuf[(row * 256 + xp) * PITCH + ci] = f32_to_bf16(v);
  }
  __syncthreads();

  // ---- A fragments: one per tap, kept in VGPRs for all tiles ----
  // A 16x32, this code's contiguous k order: lane(M)=col, half=grp; VGPR v: K = 16g+2v (matches B)
  v16bf afrag[9];
  #pragma unroll
  for (int t9 = 0; t9 < 9; ++t9) {
    v8u au;
    #pragma unroll
    for (int v = 0; v < 8; ++v) {
      const int kb = 16 * grp + 2 * v;
      au[v] = *(const unsigned int*)&wbuf[(t9 * 16 + col) * 32 + kb];
    }
    afrag[t9] = __builtin_bit_cast(v16bf, au);
  }

  // per-lane bias/BN constants: C layout -> VGPR r holds co = 8*grp + r
  float bv[8];
  #pragma unroll
  for (int r = 0; r < 8; ++r) bv[r] = beta2[b * 16 + 8 * grp + r];

  const int cbase = 16 * grp;            // B-layout K base for this lane half

  // ---- 64 tiles (2 rows x 32 col-tiles) over 8 waves ----
  for (int t = wave; t < 64; t += 8) {
    const int y    = y0 + (t >> 5);
    const int tile = t & 31;
    const int x0   = tile * 16;
    const int px   = x0 + col;

    // The three kw taps touch only TWO distinct source columns after the
    // nearest-upsample halving; kw=1 aliases one of them by parity of px.
    const bool ok0 = (px >= 1);          // left neighbor in-range
    const bool ok1 = (px < 511);         // right neighbor in-range
    const int xs0 = (ok0 ? px - 1 : 0) >> 1;
    const int xs1 = (ok1 ? px + 1 : 511) >> 1;
    const bool odd = (px & 1) != 0;

    v8f c = {};
    #pragma unroll
    for (int kh = 0; kh < 3; ++kh) {
      const int yy = y + kh - 1;                 // upsampled row, zero-pad OOB
      if (yy < 0 || yy > 511) continue;
      const int rowsel = (yy >> 1) - rbase;      // 0..2
      const unsigned rb = (unsigned)(rowsel * 256);
      const unsigned off0 = (rb + xs0) * PITCH + cbase;  // halfword index
      const unsigned off1 = (rb + xs1) * PITCH + cbase;
      // B 32x16, contiguous k order (same as A): lane(N)=col, half=grp; VGPR v holds K=16g+2v,+1
      // consecutive-channel pair => one aligned b32 LDS read, imm offsets +4v bytes
      v8u L0, L1;
      #pragma unroll
      for (int v = 0; v < 8; ++v) {
        L0[v] = *(const unsigned int*)&sbuf[off0 + 2 * v];
        L1[v] = *(const unsigned int*)&sbuf[off1 + 2 * v];
      }
      v8u B0, Bm, B2;
      #pragma unroll
      for (int v = 0; v < 8; ++v) {
        B0[v] = ok0 ? L0[v] : 0u;
        Bm[v] = odd ? L0[v] : L1[v];
        B2[v] = ok1 ? L1[v] : 0u;
      }
      c = __builtin_amdgcn_wmma_f32_16x16x32_bf16(
              false, afrag[kh * 3 + 0], false, __builtin_bit_cast(v16bf, B0),
              (short)0, c, false, false);
      c = __builtin_amdgcn_wmma_f32_16x16x32_bf16(
              false, afrag[kh * 3 + 1], false, __builtin_bit_cast(v16bf, Bm),
              (short)0, c, false, false);
      c = __builtin_amdgcn_wmma_f32_16x16x32_bf16(
              false, afrag[kh * 3 + 2], false, __builtin_bit_cast(v16bf, B2),
              (short)0, c, false, false);
    }
    // ---- epilogue: + (bias*scale + shift), ReLU, store ----
    #pragma unroll
    for (int r = 0; r < 8; ++r) {
      const int co = 8 * grp + r;
      float vv = c[r] + bv[r];
      vv = vv > 0.f ? vv : 0.f;
      out[(((size_t)(b * 16 + co) * 512) + y) * 512 + x0 + col] = vv;
    }
  }
}

// ---------------------------------------------------------------------------
extern "C" void kernel_launch(void* const* d_in, const int* in_sizes, int n_in,
                              void* d_out, int out_size, void* d_ws, size_t ws_size,
                              hipStream_t stream) {
  const float* x      = (const float*)d_in[0];
  const float* weight = (const float*)d_in[1];
  const float* bias   = (const float*)d_in[2];
  const float* fc1_w  = (const float*)d_in[3];
  const float* fc2_w  = (const float*)d_in[4];
  const float* fc2_b  = (const float*)d_in[5];
  const float* gamma  = (const float*)d_in[6];
  const float* beta   = (const float*)d_in[7];
  const float* mean   = (const float*)d_in[8];
  const float* var    = (const float*)d_in[9];
  float* out = (float*)d_out;

  char* ws = (char*)d_ws;
  float*          pooled = (float*)ws;                   // 256 f
  float*          beta2  = (float*)(ws + 1024);          // 128 f
  unsigned short* aggW   = (unsigned short*)(ws + 2048); // 8*4608 bf16, 8B-aligned chunks

  k_pool<<<256, 256, 0, stream>>>(x, pooled);
  k_attn_agg<<<8, 256, 0, stream>>>(pooled, weight, bias, fc1_w, fc2_w, fc2_b,
                                    gamma, beta, mean, var, aggW, beta2);
  k_conv_wmma<<<dim3(256, 8), 256, 0, stream>>>(x, aggW, beta2, out);
}
